// ComplexMamba3LayerV8_7816840479013
// MI455X (gfx1250) — matrix operand order repaired and checked by reading, not compiled, not run on hardware
//
#include <hip/hip_runtime.h>
#include <hip/hip_bf16.h>
#include <math.h>

// ---------------------------------------------------------------------------
// ComplexMamba3LayerV8 on MI455X (gfx1250, wave32, WMMA)
// GEMMs: v_wmma_f32_16x16x32_bf16, fragment-major LDS staging (frag load =
// two ds_load_b128), register-batched global loads, packed v_cvt_pk_bf16_f32
// conversion, double-buffered LDS so staging overlaps WMMA.
// ---------------------------------------------------------------------------

typedef __attribute__((ext_vector_type(16))) __bf16 v16bf;
typedef __attribute__((ext_vector_type(2)))  __bf16 v2bf;
typedef __attribute__((ext_vector_type(8)))  float  v8f;

#define Bn 4
#define Sn 2048
#define Dn 512
#define Nn 16
#define Mrows (Bn * Sn)          // 8192
#define SCALE_C 1.0f
#define GAMMA_C 0.01f

__device__ __forceinline__ unsigned short f2bf(float f) {
  return __builtin_bit_cast(unsigned short, (__bf16)f);   // hw cvt, RTE
}
__device__ __forceinline__ unsigned pack2(float lo, float hi) {
  v2bf p; p[0] = (__bf16)lo; p[1] = (__bf16)hi;           // v_cvt_pk_bf16_f32
  return __builtin_bit_cast(unsigned, p);
}

// Fragment-major LDS layout:
// A tile (rows x 32k): slot = (rowSubtile*32 + lane)*16 + j
//   ISA A map: K=(j&7)+(j>=8?16:0)+8*khalf, lane=(r&15)+16*khalf
//   inverse:   khalf=(k>>3)&1, j=(k&7)+(k>=16?8:0)  (even k pairs contiguous)
// B tile (32k x cols): slot = (colSubtile*32 + lane)*16 + j
//   ISA B map: K=(j&7)+(j>=8?16:0)+8*khalf, lane=(col&15)+16*khalf
//   inverse:   lane=(col&15)+16*((k>>3)&1), j=(k&7)+(k>=16?8:0)       (even k pairs contiguous)

// -------------------- complex GEMM: C = A @ B (complex), bf16 WMMA ----------
// Block tile 128(M) x 64(N), K-step 32, double-buffered LDS. 8 waves 4x2,
// each wave 2x2 subtiles, 3 accumulators/subtile (Re+, Re-, Im).
__global__ __launch_bounds__(256)
void cgemm_wmma(const float* __restrict__ Are, const float* __restrict__ Aim,
                const float* __restrict__ Bre, const float* __restrict__ Bim,
                float* __restrict__ Cre, float* __restrict__ Cim,
                const float* __restrict__ colScale, int Ksz, int Nsz) {
  __shared__ __align__(32) unsigned short sAr[2][8 * 32 * 16];
  __shared__ __align__(32) unsigned short sAi[2][8 * 32 * 16];
  __shared__ __align__(32) unsigned short sBr[2][4 * 32 * 16];
  __shared__ __align__(32) unsigned short sBi[2][4 * 32 * 16];

  const int tid  = threadIdx.x;
  const int lane = tid & 31;
  const int w    = tid >> 5;
  const int wm   = w & 3, wn = w >> 2;
  const int m0   = blockIdx.x * 128;
  const int n0   = blockIdx.y * 64;
  const int khalf = lane >> 4;
  const int colB  = lane & 15;

  float2 nAr[8], nAi[8];       // prefetched A pairs (even k)
  float2 nBr[4], nBi[4];       // prefetched B k-row pairs

  auto loadTile = [&](int kt) {
#pragma unroll
    for (int i = 0; i < 8; ++i) {
      int p = i * 256 + tid;            // pair index 0..2047
      int r = p >> 4;                   // row 0..127
      int c = (p & 15) * 2;             // even k
      size_t g = (size_t)(m0 + r) * Ksz + kt + c;
      nAr[i] = *(const float2*)(Are + g);
      nAi[i] = *(const float2*)(Aim + g);
    }
#pragma unroll
    for (int i = 0; i < 4; ++i) {
      int p = i * 256 + tid;            // 0..1023
      int c = p & 63;
      int r = (p >> 6) * 2;             // even k row
      size_t g = (size_t)(kt + r) * Nsz + n0 + c;
      nBr[i].x = Bre[g];       nBi[i].x = Bim[g];
      nBr[i].y = Bre[g + Nsz]; nBi[i].y = Bim[g + Nsz];
    }
  };
  auto storeTile = [&](int buf) {
    unsigned* wr = (unsigned*)sAr[buf];
    unsigned* wi = (unsigned*)sAi[buf];
    unsigned* br = (unsigned*)sBr[buf];
    unsigned* bi = (unsigned*)sBi[buf];
#pragma unroll
    for (int i = 0; i < 8; ++i) {
      int p = i * 256 + tid;
      int r = p >> 4;
      int c = (p & 15) * 2;
      int kh = (c >> 3) & 1;
      int j0 = (c & 7) + ((c >= 16) ? 8 : 0);
      int slot = ((r >> 4) * 32 + (r & 15) + 16 * kh) * 8 + (j0 >> 1);
      wr[slot] = pack2(nAr[i].x, nAr[i].y);
      wi[slot] = pack2(nAi[i].x, nAi[i].y);
    }
#pragma unroll
    for (int i = 0; i < 4; ++i) {
      int p = i * 256 + tid;
      int c = p & 63;
      int r = (p >> 6) * 2;
      int slot = ((c >> 4) * 32 + (c & 15) + 16 * ((r >> 3) & 1)) * 8 + (((r & 7) + ((r >= 16) ? 8 : 0)) >> 1);
      br[slot] = pack2(nBr[i].x, nBr[i].y);
      bi[slot] = pack2(nBi[i].x, nBi[i].y);
    }
  };

  const v8f vzero = {};
  v8f accRp[2][2], accRn[2][2], accI[2][2];
#pragma unroll
  for (int mi = 0; mi < 2; ++mi)
#pragma unroll
    for (int ni = 0; ni < 2; ++ni) {
      accRp[mi][ni] = vzero; accRn[mi][ni] = vzero; accI[mi][ni] = vzero;
    }

  loadTile(0);
  storeTile(0);
  __syncthreads();

  for (int kt = 0; kt < Ksz; kt += 32) {
    const int cur = (kt >> 5) & 1;
    const bool more = (kt + 32 < Ksz);
    if (more) loadTile(kt + 32);        // overlap global loads with WMMA

    const v16bf* fAr = (const v16bf*)sAr[cur];
    const v16bf* fAi = (const v16bf*)sAi[cur];
    const v16bf* fBr = (const v16bf*)sBr[cur];
    const v16bf* fBi = (const v16bf*)sBi[cur];

    v16bf aR[2], aI[2], bR[2], bI[2];
#pragma unroll
    for (int mi = 0; mi < 2; ++mi) {
      aR[mi] = fAr[(wm * 2 + mi) * 32 + lane];   // two ds_load_b128
      aI[mi] = fAi[(wm * 2 + mi) * 32 + lane];
    }
#pragma unroll
    for (int ni = 0; ni < 2; ++ni) {
      bR[ni] = fBr[(wn * 2 + ni) * 32 + lane];
      bI[ni] = fBi[(wn * 2 + ni) * 32 + lane];
    }
#pragma unroll
    for (int mi = 0; mi < 2; ++mi)
#pragma unroll
      for (int ni = 0; ni < 2; ++ni) {
        accRp[mi][ni] = __builtin_amdgcn_wmma_f32_16x16x32_bf16(
            false, aR[mi], false, bR[ni], (short)0, accRp[mi][ni], false, false);
        accRn[mi][ni] = __builtin_amdgcn_wmma_f32_16x16x32_bf16(
            false, aI[mi], false, bI[ni], (short)0, accRn[mi][ni], false, false);
        accI[mi][ni] = __builtin_amdgcn_wmma_f32_16x16x32_bf16(
            false, aR[mi], false, bI[ni], (short)0, accI[mi][ni], false, false);
        accI[mi][ni] = __builtin_amdgcn_wmma_f32_16x16x32_bf16(
            false, aI[mi], false, bR[ni], (short)0, accI[mi][ni], false, false);
      }
    if (more) storeTile(cur ^ 1);
    __syncthreads();
  }

#pragma unroll
  for (int mi = 0; mi < 2; ++mi)
#pragma unroll
    for (int ni = 0; ni < 2; ++ni)
#pragma unroll
      for (int r = 0; r < 8; ++r) {
        int mm = m0 + wm * 32 + mi * 16 + r + 8 * khalf;     // ISA C/D layout
        int nn = n0 + wn * 32 + ni * 16 + colB;
        float sc = colScale ? colScale[nn] : 1.0f;
        Cre[(size_t)mm * Nsz + nn] = (accRp[mi][ni][r] - accRn[mi][ni][r]) * sc;
        Cim[(size_t)mm * Nsz + nn] = accI[mi][ni][r] * sc;
      }
}

// -------------------- real GEMM with softplus(x + bias) epilogue ------------
__global__ __launch_bounds__(256)
void rgemm_dt(const float* __restrict__ A, const float* __restrict__ Bm,
              const float* __restrict__ bias, float* __restrict__ C,
              int Ksz, int Nsz) {
  __shared__ __align__(32) unsigned short sA[2][8 * 32 * 16];
  __shared__ __align__(32) unsigned short sB[2][4 * 32 * 16];
  const int tid = threadIdx.x, lane = tid & 31, w = tid >> 5;
  const int wm = w & 3, wn = w >> 2;
  const int m0 = blockIdx.x * 128, n0 = blockIdx.y * 64;
  const int khalf = lane >> 4, colB = lane & 15;

  float2 nA[8], nB[4];
  auto loadTile = [&](int kt) {
#pragma unroll
    for (int i = 0; i < 8; ++i) {
      int p = i * 256 + tid;
      int r = p >> 4, c = (p & 15) * 2;
      nA[i] = *(const float2*)(A + (size_t)(m0 + r) * Ksz + kt + c);
    }
#pragma unroll
    for (int i = 0; i < 4; ++i) {
      int p = i * 256 + tid;
      int c = p & 63;
      int r = (p >> 6) * 2;
      size_t g = (size_t)(kt + r) * Nsz + n0 + c;
      nB[i].x = Bm[g]; nB[i].y = Bm[g + Nsz];
    }
  };
  auto storeTile = [&](int buf) {
    unsigned* wa = (unsigned*)sA[buf];
    unsigned* wb = (unsigned*)sB[buf];
#pragma unroll
    for (int i = 0; i < 8; ++i) {
      int p = i * 256 + tid;
      int r = p >> 4, c = (p & 15) * 2;
      int kh = (c >> 3) & 1;
      int j0 = (c & 7) + ((c >= 16) ? 8 : 0);
      wa[((r >> 4) * 32 + (r & 15) + 16 * kh) * 8 + (j0 >> 1)] = pack2(nA[i].x, nA[i].y);
    }
#pragma unroll
    for (int i = 0; i < 4; ++i) {
      int p = i * 256 + tid;
      int c = p & 63;
      int r = (p >> 6) * 2;
      wb[((c >> 4) * 32 + (c & 15) + 16 * ((r >> 3) & 1)) * 8 + (((r & 7) + ((r >= 16) ? 8 : 0)) >> 1)] =
          pack2(nB[i].x, nB[i].y);
    }
  };

  const v8f vzero = {};
  v8f acc[2][2];
#pragma unroll
  for (int mi = 0; mi < 2; ++mi)
#pragma unroll
    for (int ni = 0; ni < 2; ++ni) acc[mi][ni] = vzero;

  loadTile(0);
  storeTile(0);
  __syncthreads();

  for (int kt = 0; kt < Ksz; kt += 32) {
    const int cur = (kt >> 5) & 1;
    const bool more = (kt + 32 < Ksz);
    if (more) loadTile(kt + 32);
    const v16bf* fA = (const v16bf*)sA[cur];
    const v16bf* fB = (const v16bf*)sB[cur];
    v16bf af[2], bf[2];
#pragma unroll
    for (int mi = 0; mi < 2; ++mi) af[mi] = fA[(wm * 2 + mi) * 32 + lane];
#pragma unroll
    for (int ni = 0; ni < 2; ++ni) bf[ni] = fB[(wn * 2 + ni) * 32 + lane];
#pragma unroll
    for (int mi = 0; mi < 2; ++mi)
#pragma unroll
      for (int ni = 0; ni < 2; ++ni)
        acc[mi][ni] = __builtin_amdgcn_wmma_f32_16x16x32_bf16(
            false, af[mi], false, bf[ni], (short)0, acc[mi][ni], false, false);
    if (more) storeTile(cur ^ 1);
    __syncthreads();
  }
#pragma unroll
  for (int mi = 0; mi < 2; ++mi)
#pragma unroll
    for (int ni = 0; ni < 2; ++ni)
#pragma unroll
      for (int r = 0; r < 8; ++r) {
        int mm = m0 + wm * 32 + mi * 16 + r + 8 * khalf;
        int nn = n0 + wn * 32 + ni * 16 + colB;
        float x = acc[mi][ni][r] + bias[nn];
        C[(size_t)mm * Nsz + nn] = (x > 20.0f) ? x : log1pf(__expf(x));
      }
}

// -------------------- skinny GEMMs: Bm = xr@W_B, Cm = xr@W_C (N=16) ---------
__global__ __launch_bounds__(256)
void gemm_bc(const float* __restrict__ A, const float* __restrict__ WB,
             const float* __restrict__ WC, float* __restrict__ Bo,
             float* __restrict__ Co, int Ksz) {
  __shared__ __align__(32) unsigned short sA[8 * 32 * 16];
  __shared__ __align__(32) unsigned short sB[32 * 16];
  __shared__ __align__(32) unsigned short sC[32 * 16];
  const int tid = threadIdx.x, lane = tid & 31, w = tid >> 5;
  const int m0 = blockIdx.x * 128;
  const int khalf = lane >> 4, colB = lane & 15;
  unsigned* wA = (unsigned*)sA;
  unsigned* wB32 = (unsigned*)sB;
  unsigned* wC32 = (unsigned*)sC;
  const v16bf* fA = (const v16bf*)sA;
  const v16bf* fB = (const v16bf*)sB;
  const v16bf* fC = (const v16bf*)sC;
  const v8f vzero = {};
  v8f accB = vzero, accC = vzero;

  for (int kt = 0; kt < Ksz; kt += 32) {
    float2 nA[8];
#pragma unroll
    for (int i = 0; i < 8; ++i) {                 // batch global loads
      int p = i * 256 + tid;
      int r = p >> 4, c = (p & 15) * 2;
      nA[i] = *(const float2*)(A + (size_t)(m0 + r) * Ksz + kt + c);
    }
    // W_B/W_C: 32x16, one k-row pair per thread
    int cB = tid & 15;
    int rB = (tid >> 4) * 2;
    float2 vb, vc;
    vb.x = WB[(kt + rB) * Nn + cB]; vb.y = WB[(kt + rB + 1) * Nn + cB];
    vc.x = WC[(kt + rB) * Nn + cB]; vc.y = WC[(kt + rB + 1) * Nn + cB];
    if (kt + 32 < Ksz)                             // gfx1250 global_prefetch_b8
      __builtin_prefetch(&A[(size_t)(m0 + (tid >> 1)) * Ksz + kt + 32], 0, 1);
#pragma unroll
    for (int i = 0; i < 8; ++i) {
      int p = i * 256 + tid;
      int r = p >> 4, c = (p & 15) * 2;
      int kh = (c >> 3) & 1;
      int j0 = (c & 7) + ((c >= 16) ? 8 : 0);
      wA[((r >> 4) * 32 + (r & 15) + 16 * kh) * 8 + (j0 >> 1)] = pack2(nA[i].x, nA[i].y);
    }
    {
      int slot = ((cB & 15) + 16 * ((rB >> 3) & 1)) * 8 + (((rB & 7) + ((rB >= 16) ? 8 : 0)) >> 1);
      wB32[slot] = pack2(vb.x, vb.y);
      wC32[slot] = pack2(vc.x, vc.y);
    }
    __syncthreads();
    v16bf af = fA[w * 32 + lane];
    v16bf bB = fB[lane];
    v16bf bC = fC[lane];
    accB = __builtin_amdgcn_wmma_f32_16x16x32_bf16(false, af, false, bB,
                                                   (short)0, accB, false, false);
    accC = __builtin_amdgcn_wmma_f32_16x16x32_bf16(false, af, false, bC,
                                                   (short)0, accC, false, false);
    __syncthreads();
  }
#pragma unroll
  for (int r = 0; r < 8; ++r) {
    int mm = m0 + w * 16 + r + 8 * khalf;
    Bo[(size_t)mm * Nn + colB] = accB[r];
    Co[(size_t)mm * Nn + colB] = accC[r];
  }
}

// -------------------- causal depthwise conv (K=4) + modSiLU -----------------
__global__ void conv_gate(const float* __restrict__ UR, const float* __restrict__ UI,
                          const float* __restrict__ cw,
                          float* __restrict__ GR, float* __restrict__ GI) {
  int idx = blockIdx.x * blockDim.x + threadIdx.x;
  if (idx >= Bn * Sn * Dn) return;
  int d = idx % Dn;
  int s = (idx / Dn) % Sn;
  int b = idx / (Dn * Sn);
  float r = 0.f, im = 0.f;
#pragma unroll
  for (int k = 0; k < 4; ++k) {
    int ss = s + k - 3;
    if (ss >= 0) {
      float wv = cw[k * Dn + d];
      size_t g = ((size_t)b * Sn + ss) * Dn + d;
      r += wv * UR[g];
      im += wv * UI[g];
    }
  }
  float mag = sqrtf(r * r + im * im);
  float sg = 1.0f / (1.0f + __expf(-mag));
  GR[idx] = r * sg;
  GI[idx] = im * sg;
}

// -------------------- sequential selective scan -----------------------------
__global__ __launch_bounds__(256)
void scan_kernel(const float* __restrict__ DT, const float* __restrict__ BM,
                 const float* __restrict__ CM, const float* __restrict__ GR,
                 const float* __restrict__ GI, const float* __restrict__ A_log,
                 const float* __restrict__ D_skip,
                 float* __restrict__ YR, float* __restrict__ YI) {
  const int b = blockIdx.x >> 1;
  const int d = ((blockIdx.x & 1) << 8) + threadIdx.x;
  float Ac[Nn];
#pragma unroll
  for (int n = 0; n < Nn; ++n) Ac[n] = -__expf(A_log[d * Nn + n]);
  const float dsk = D_skip[d];
  float hr[Nn] = {}, hi[Nn] = {};
  __shared__ float sB[Nn], sC[Nn];
  for (int s = 0; s < Sn; ++s) {
    if (threadIdx.x < Nn)
      sB[threadIdx.x] = BM[((size_t)b * Sn + s) * Nn + threadIdx.x];
    else if (threadIdx.x < 2 * Nn)
      sC[threadIdx.x - Nn] = CM[((size_t)b * Sn + s) * Nn + (threadIdx.x - Nn)];
    __syncthreads();
    const size_t g = ((size_t)b * Sn + s) * Dn + d;
    const float dt = DT[g];
    const float ur = GR[g], ui = GI[g];
    const float dur = dt * ur, dui = dt * ui;
    float yr = 0.f, yi = 0.f;
#pragma unroll
    for (int n = 0; n < Nn; ++n) {
      const float dec = __expf(dt * Ac[n]);
      hr[n] = dec * hr[n] + dur * sB[n];
      hi[n] = dec * hi[n] + dui * sB[n];
      yr += hr[n] * sC[n];
      yi += hi[n] * sC[n];
    }
    YR[g] = yr + dsk * ur;
    YI[g] = yi + dsk * ui;
    __syncthreads();
  }
}

// -------------------- wrapper pieces ----------------------------------------
__global__ void lcoef_kernel(const float* __restrict__ Lr, const float* __restrict__ Li,
                             float* __restrict__ LC) {
  int d = blockIdx.x * blockDim.x + threadIdx.x;
  if (d >= Dn) return;
  float s = 0.f;
#pragma unroll
  for (int k = 0; k < 4; ++k) {
    float a = Lr[k * Dn + d], b = Li[k * Dn + d];
    s += a * a + b * b;
  }
  LC[d] = 1.0f - GAMMA_C * s;
}

__global__ __launch_bounds__(256)
void diag_kernel(const float* __restrict__ BR, const float* __restrict__ BI,
                 float* __restrict__ DIAG) {
  int t = blockIdx.x * blockDim.x + threadIdx.x;   // 0..B*D-1
  if (t >= Bn * Dn) return;
  int b = t / Dn, d = t % Dn;
  float acc = 0.f;
  for (int s = 0; s < Sn; ++s) {
    size_t g = ((size_t)b * Sn + s) * Dn + d;
    float br = BR[g], bi = BI[g];
    acc += br * br + bi * bi;                      // |branch|^2 = mag^2
  }
  DIAG[t] = acc / (float)Sn;
}

__global__ void cond_kernel(const float* __restrict__ DIAG, float* __restrict__ ESC) {
  __shared__ float smx[256], smn[256];
  int b = blockIdx.x, t = threadIdx.x;
  float mx = -1e30f, mn = 1e30f;
  for (int d = t; d < Dn; d += 256) {
    float v = DIAG[b * Dn + d];
    mx = fmaxf(mx, v);
    mn = fminf(mn, v);
  }
  smx[t] = mx; smn[t] = mn;
  __syncthreads();
  for (int off = 128; off > 0; off >>= 1) {
    if (t < off) {
      smx[t] = fmaxf(smx[t], smx[t + off]);
      smn[t] = fminf(smn[t], smn[t + off]);
    }
    __syncthreads();
  }
  if (t == 0) ESC[b] = (smx[0] / (smn[0] + 1e-8f) > 100.0f) ? 1.0f : 0.0f;
}

__global__ void final_kernel(const float* __restrict__ XR, const float* __restrict__ XI,
                             const float* __restrict__ BR, const float* __restrict__ BI,
                             const float* __restrict__ phi, const float* __restrict__ ESC,
                             const float* __restrict__ ilog, float* __restrict__ OUT) {
  const int total = Bn * Sn * Dn;
  int idx = blockIdx.x * blockDim.x + threadIdx.x;
  if (idx >= total) return;
  int d = idx % Dn;
  int b = idx / (Dn * Sn);
  float br = BR[idx], bi = BI[idx];
  if (ESC[b] != 0.0f) {
    float pr = cosf(phi[d]), pi = sinf(phi[d]);
    float nr = br * pr - bi * pi;
    float ni = br * pi + bi * pr;
    br = nr; bi = ni;
  }
  br *= SCALE_C; bi *= SCALE_C;
  // 1x1 Sinkhorn is exactly 1 -> coef = exp(i * imag_logit)
  float il = ilog[0];
  float cr = cosf(il), ci = sinf(il);
  OUT[idx]         = XR[idx] + cr * br - ci * bi;
  OUT[total + idx] = XI[idx] + cr * bi + ci * br;
}

// ---------------------------------------------------------------------------
extern "C" void kernel_launch(void* const* d_in, const int* in_sizes, int n_in,
                              void* d_out, int out_size, void* d_ws, size_t ws_size,
                              hipStream_t stream) {
  const float* XR     = (const float*)d_in[0];
  const float* XI     = (const float*)d_in[1];
  const float* Win_r  = (const float*)d_in[2];
  const float* Win_i  = (const float*)d_in[3];
  const float* conv_w = (const float*)d_in[4];
  const float* W_dt   = (const float*)d_in[5];
  const float* b_dt   = (const float*)d_in[6];
  const float* A_log  = (const float*)d_in[7];
  const float* W_B    = (const float*)d_in[8];
  const float* W_C    = (const float*)d_in[9];
  const float* D_skip = (const float*)d_in[10];
  const float* Wout_r = (const float*)d_in[11];
  const float* Wout_i = (const float*)d_in[12];
  const float* L_r    = (const float*)d_in[13];
  const float* L_i    = (const float*)d_in[14];
  const float* phi    = (const float*)d_in[15];
  const float* Hlogi  = (const float*)d_in[17];
  float* OUT = (float*)d_out;

  float* ws = (float*)d_ws;
  const size_t MD = (size_t)Mrows * Dn;            // 8192*512
  float* UR = ws;            float* UI = UR + MD;  // in-proj out, reused as Y
  float* GR = UI + MD;       float* GI = GR + MD;  // gated u, reused as branch
  float* DT = GI + MD;
  float* BM = DT + MD;       float* CM = BM + (size_t)Mrows * Nn;
  float* DIAG = CM + (size_t)Mrows * Nn;
  float* ESC  = DIAG + Bn * Dn;
  float* LC   = ESC + 8;

  const int total = Bn * Sn * Dn;
  dim3 gemmGrid(Mrows / 128, Dn / 64);             // 64 x 8

  // 1) Lindblad column scale
  lcoef_kernel<<<2, 256, 0, stream>>>(L_r, L_i, LC);
  // 2) in-projection (complex)
  cgemm_wmma<<<gemmGrid, 256, 0, stream>>>(XR, XI, Win_r, Win_i, UR, UI,
                                           nullptr, Dn, Dn);
  // 3) causal conv + modSiLU gate
  conv_gate<<<(total + 255) / 256, 256, 0, stream>>>(UR, UI, conv_w, GR, GI);
  // 4) dt projection + softplus
  rgemm_dt<<<gemmGrid, 256, 0, stream>>>(GR, W_dt, b_dt, DT, Dn, Dn);
  // 5) B/C projections
  gemm_bc<<<Mrows / 128, 256, 0, stream>>>(GR, W_B, W_C, BM, CM, Dn);
  // 6) selective scan (writes Y into UR/UI, which are free now)
  scan_kernel<<<Bn * (Dn / 256), 256, 0, stream>>>(DT, BM, CM, GR, GI, A_log,
                                                   D_skip, UR, UI);
  // 7) out-projection (complex) with Lindblad scale -> branch in GR/GI
  cgemm_wmma<<<gemmGrid, 256, 0, stream>>>(UR, UI, Wout_r, Wout_i, GR, GI,
                                           LC, Dn, Dn);
  // 8) diagonal of the Gram matrix (only part the reference ever uses)
  diag_kernel<<<(Bn * Dn + 255) / 256, 256, 0, stream>>>(GR, GI, DIAG);
  // 9) condition / escape flags
  cond_kernel<<<Bn, 256, 0, stream>>>(DIAG, ESC);
  // 10) rotation + residual combine -> [2,B,S,D]
  final_kernel<<<(total + 255) / 256, 256, 0, stream>>>(XR, XI, GR, GI, phi,
                                                        ESC, Hlogi, OUT);
}
